// DecoderGroupedQueryHeadAttentionRope_55061480735461
// MI455X (gfx1250) — matrix operand order repaired and checked by reading, not compiled, not run on hardware
//
#include <hip/hip_runtime.h>
#include <hip/hip_bf16.h>
#include <math.h>

#define TB 2
#define TT 2048
#define TC 2048
#define NH 16
#define NKV 4
#define HD 128
#define BT (TB*TT)

typedef __attribute__((ext_vector_type(16))) _Float16 v16h;
typedef __attribute__((ext_vector_type(8)))  float    v8f;

union AF16 { v16h v; uint4 q[2]; };

// ---------------------------------------------------------------- conversion
__global__ void k_f32_to_f16(const float* __restrict__ in, _Float16* __restrict__ out, int n) {
  int i = blockIdx.x * blockDim.x + threadIdx.x;
  if (i < n) out[i] = (_Float16)in[i];
}

// ---------------------------------------------------------------- GEMM: out[m,n] = sum_k A[m,k]*W[n,k]
// A: [M,K] f16 row-major.  W: [N,K] f16 row-major.  Block tile 128x128, 8 waves.
template<bool WRITE_F32>
__global__ __launch_bounds__(256)
void k_gemm_awt(const _Float16* __restrict__ A, const _Float16* __restrict__ W,
                _Float16* __restrict__ outh, float* __restrict__ outf,
                const float* __restrict__ bias, int N, int K)
{
  __shared__ _Float16 lb[128 * 32];      // [n_local][k] tile of W
  const int tid  = threadIdx.x;
  const int wave = tid >> 5;
  const int lane = tid & 31;
  const int g    = lane >> 4;            // half-wave group
  const int ln   = lane & 15;
  const int m0   = blockIdx.y * 128 + wave * 16;
  const int n0   = blockIdx.x * 128;

  v8f acc[8] = {};
  for (int k0 = 0; k0 < K; k0 += 32) {
    __syncthreads();
    { // stage W tile: 128 rows x 32 halves
      int r  = tid >> 1;
      int kp = (tid & 1) << 4;
      const uint4* s4 = (const uint4*)(W + (size_t)(n0 + r) * K + k0 + kp);
      uint4* d4 = (uint4*)(lb + r * 32 + kp);
      d4[0] = s4[0];
      d4[1] = s4[1];
    }
    // A fragment: lane row m0+ln; halves i -> k0+g*16+i (contiguous)
    AF16 af;
    const _Float16* arow = A + (size_t)(m0 + ln) * K + k0 + g * 16;
    af.q[0] = *(const uint4*)(arow);
    af.q[1] = *(const uint4*)(arow + 8);
    __syncthreads();
#pragma unroll
    for (int nt = 0; nt < 8; ++nt) {
      // B fragment: lane col n0+nt*16+ln; halves i -> k = g*16+i (contiguous in lb row)
      AF16 bf;
      const uint4* b4 = (const uint4*)(lb + (nt * 16 + ln) * 32 + g * 16);
      bf.q[0] = b4[0];
      bf.q[1] = b4[1];
      acc[nt] = __builtin_amdgcn_wmma_f32_16x16x32_f16(false, af.v, false, bf.v,
                                                       (short)0, acc[nt], false, false);
    }
  }
#pragma unroll
  for (int nt = 0; nt < 8; ++nt) {
    int n = n0 + nt * 16 + ln;
#pragma unroll
    for (int r = 0; r < 8; ++r) {
      int m = m0 + r + 8 * g;            // C layout: row = vgpr + 8*(lane/16)
      float v = acc[nt][r];
      if constexpr (WRITE_F32) outf[(size_t)m * N + n] = v + bias[n];
      else                     outh[(size_t)m * N + n] = (_Float16)v;
    }
  }
}

// ---------------------------------------------------------------- RoPE + head packing
// lin: [B,T,row_cols] f16 (projection output). out: [B,heads,T,HD] f16.
__global__ void k_rope_pack(const _Float16* __restrict__ lin, const float* __restrict__ fr,
                            _Float16* __restrict__ out, int heads, int col_off, int row_cols)
{
  int i = blockIdx.x * blockDim.x + threadIdx.x;
  int total = TB * heads * TT * (HD / 2);
  if (i >= total) return;
  int j = i & 63;  int r = i >> 6;
  int t = r & (TT - 1);  r >>= 11;
  int h = r % heads;  int b = r / heads;
  float c = fr[((size_t)t * 64 + j) * 2 + 0];
  float s = fr[((size_t)t * 64 + j) * 2 + 1];
  const _Float16* src = lin + ((size_t)b * TT + t) * row_cols + col_off + h * HD + 2 * j;
  float x0 = (float)src[0], x1 = (float)src[1];
  _Float16* dst = out + ((size_t)(b * heads + h) * TT + t) * HD + 2 * j;
  dst[0] = (_Float16)(x0 * c - x1 * s);
  dst[1] = (_Float16)(x0 * s + x1 * c);
}

__global__ void k_pack_v(const _Float16* __restrict__ lin, _Float16* __restrict__ out)
{
  int i = blockIdx.x * blockDim.x + threadIdx.x;  // B*NKV*TT*HD
  if (i >= TB * NKV * TT * HD) return;
  int d = i & 127;  int r = i >> 7;
  int t = r & (TT - 1);  r >>= 11;
  int h = r & 3;  int b = r >> 2;
  out[((size_t)(b * NKV + h) * TT + t) * HD + d] =
      lin[((size_t)b * TT + t) * (2 * NKV * HD) + NKV * HD + h * HD + d];
}

// ---------------------------------------------------------------- flash attention
// Q: [B,NH,T,HD] f16 (roped). Kh: [B,NKV,T,HD] f16 (roped). Vh: [B,NKV,T,HD] f16.
// O: [B,T,C] f16. Block = 4 waves, wave owns 16 query rows; key blocks of 32.
__global__ __launch_bounds__(128)
void k_attn(const _Float16* __restrict__ Q, const _Float16* __restrict__ Kh,
            const _Float16* __restrict__ Vh, _Float16* __restrict__ O)
{
  __shared__ _Float16 Kb[32 * 128];   // [key][d]
  __shared__ _Float16 VbT[128 * 32];  // [d][key]
  __shared__ _Float16 Ps[4 * 16 * 32];// per-wave P staging [row][key]
  const int tid  = threadIdx.x;
  const int wave = tid >> 5;
  const int lane = tid & 31;
  const int g    = lane >> 4;
  const int ln   = lane & 15;
  const int bh   = blockIdx.y;
  const int b    = bh >> 4;
  const int h    = bh & 15;
  const int kvh  = h & 3;             // reference tiles KV heads: head h -> kv h%4
  const int q0   = blockIdx.x * 64;
  const int qm   = q0 + wave * 16;

  // Q fragments for the whole 16x128 tile (A layout)
  AF16 qf[4];
  {
    const _Float16* qrow = Q + ((size_t)(b * NH + h) * TT + qm + ln) * HD;
#pragma unroll
    for (int kc = 0; kc < 4; ++kc) {
      qf[kc].q[0] = *(const uint4*)(qrow + kc * 32 + g * 16);
      qf[kc].q[1] = *(const uint4*)(qrow + kc * 32 + g * 16 + 8);
    }
  }
  v8f oacc[8] = {};
  float mrow[8], lrow[8];
#pragma unroll
  for (int r = 0; r < 8; ++r) { mrow[r] = -1e30f; lrow[r] = 0.f; }

  const _Float16* Kbase = Kh + ((size_t)(b * NKV + kvh) * TT) * HD;
  const _Float16* Vbase = Vh + ((size_t)(b * NKV + kvh) * TT) * HD;
  const float scale = 0.08838834764831845f;   // 1/sqrt(128)

  for (int kb0 = 0; kb0 < q0 + 64; kb0 += 32) {
    __syncthreads();
    { // stage K (row-major) and V (transposed)
      int r = tid >> 2, p = tid & 3;
      const uint4* ks = (const uint4*)(Kbase + (size_t)(kb0 + r) * HD + p * 32);
      uint4* kd = (uint4*)(Kb + r * 128 + p * 32);
      kd[0] = ks[0]; kd[1] = ks[1]; kd[2] = ks[2]; kd[3] = ks[3];
      const _Float16* vs = Vbase + (size_t)(kb0 + r) * HD + p * 32;
#pragma unroll
      for (int i2 = 0; i2 < 32; ++i2)
        VbT[(p * 32 + i2) * 32 + r] = vs[i2];
    }
    __syncthreads();

    // S = Q @ K^T : two 16x16 n-tiles, K-dim = 128 in 4 chunks
    v8f sacc[2] = {};
#pragma unroll
    for (int kc = 0; kc < 4; ++kc) {
#pragma unroll
      for (int nt = 0; nt < 2; ++nt) {
        AF16 bf;
        const uint4* b4 = (const uint4*)(Kb + (nt * 16 + ln) * 128 + kc * 32 + g * 16);
        bf.q[0] = b4[0]; bf.q[1] = b4[1];
        sacc[nt] = __builtin_amdgcn_wmma_f32_16x16x32_f16(false, qf[kc].v, false, bf.v,
                                                          (short)0, sacc[nt], false, false);
      }
    }

    // online softmax; row = qm + r + 8*g lives on 16 lanes of group g at vgpr r
    const int key0 = kb0 + ln, key1 = kb0 + 16 + ln;
#pragma unroll
    for (int r = 0; r < 8; ++r) {
      int row = qm + r + 8 * g;
      float s0 = (key0 <= row) ? sacc[0][r] * scale : -1e30f;
      float s1 = (key1 <= row) ? sacc[1][r] * scale : -1e30f;
      float mx = fmaxf(s0, s1);
      mx = fmaxf(mx, __shfl_xor(mx, 1, 32));
      mx = fmaxf(mx, __shfl_xor(mx, 2, 32));
      mx = fmaxf(mx, __shfl_xor(mx, 4, 32));
      mx = fmaxf(mx, __shfl_xor(mx, 8, 32));
      float mnew  = fmaxf(mrow[r], mx);
      float alpha = __expf(mrow[r] - mnew);
      float p0 = (key0 <= row) ? __expf(s0 - mnew) : 0.f;
      float p1 = (key1 <= row) ? __expf(s1 - mnew) : 0.f;
      float ps = p0 + p1;
      ps += __shfl_xor(ps, 1, 32);
      ps += __shfl_xor(ps, 2, 32);
      ps += __shfl_xor(ps, 4, 32);
      ps += __shfl_xor(ps, 8, 32);
      lrow[r] = lrow[r] * alpha + ps;
      mrow[r] = mnew;
#pragma unroll
      for (int nt = 0; nt < 8; ++nt) oacc[nt][r] *= alpha;
      Ps[wave * 512 + (r + 8 * g) * 32 + ln]      = (_Float16)p0;
      Ps[wave * 512 + (r + 8 * g) * 32 + 16 + ln] = (_Float16)p1;
    }

    // same-wave LDS store->load; DS is in-order, wait for dscnt to be safe
    asm volatile("s_wait_dscnt 0" ::: "memory");

    // P as A-operand (16x32), V^T tiles as B-operand
    AF16 pf;
    const _Float16* pr = Ps + wave * 512 + ln * 32;
    pf.q[0] = *(const uint4*)(pr + g * 16);
    pf.q[1] = *(const uint4*)(pr + g * 16 + 8);
#pragma unroll
    for (int nt = 0; nt < 8; ++nt) {
      AF16 vf;
      const uint4* v4 = (const uint4*)(VbT + (nt * 16 + ln) * 32 + g * 16);
      vf.q[0] = v4[0]; vf.q[1] = v4[1];
      oacc[nt] = __builtin_amdgcn_wmma_f32_16x16x32_f16(false, pf.v, false, vf.v,
                                                        (short)0, oacc[nt], false, false);
    }
  }

  // normalize and write [B,T,C] f16
#pragma unroll
  for (int nt = 0; nt < 8; ++nt) {
    int d = nt * 16 + ln;
#pragma unroll
    for (int r = 0; r < 8; ++r) {
      int row = qm + r + 8 * g;
      float v = oacc[nt][r] / lrow[r];
      O[((size_t)b * TT + row) * TC + h * HD + d] = (_Float16)v;
    }
  }
}

// ---------------------------------------------------------------- launcher
extern "C" void kernel_launch(void* const* d_in, const int* in_sizes, int n_in,
                              void* d_out, int out_size, void* d_ws, size_t ws_size,
                              hipStream_t stream)
{
  const float* x   = (const float*)d_in[0];
  const float* fr  = (const float*)d_in[1];
  const float* wq  = (const float*)d_in[2];
  const float* wkv = (const float*)d_in[3];
  const float* wo  = (const float*)d_in[4];
  const float* bo  = (const float*)d_in[5];
  float* out = (float*)d_out;
  char* ws = (char*)d_ws;
  const size_t MB = 1024 * 1024;

  _Float16* xh    = (_Float16*)(ws + 0);        // 16 MiB
  _Float16* wqh   = (_Float16*)(ws + 16 * MB);  //  8 MiB
  _Float16* wkvh  = (_Float16*)(ws + 24 * MB);  //  4 MiB
  _Float16* woh   = (_Float16*)(ws + 28 * MB);  //  8 MiB
  _Float16* qlin  = (_Float16*)(ws + 36 * MB);  // 16 MiB
  _Float16* kvlin = (_Float16*)(ws + 52 * MB);  //  8 MiB
  _Float16* qh    = (_Float16*)(ws + 0);        // alias xh (dead after GEMMs)
  _Float16* kh    = (_Float16*)(ws + 60 * MB);  //  4 MiB
  _Float16* vh    = (_Float16*)(ws + 64 * MB);  //  4 MiB
  _Float16* ao    = (_Float16*)(ws + 36 * MB);  // alias qlin (dead after rope)

  int n;
  n = BT * TC;          k_f32_to_f16<<<(n + 255) / 256, 256, 0, stream>>>(x, xh, n);
  n = TC * TC;          k_f32_to_f16<<<(n + 255) / 256, 256, 0, stream>>>(wq, wqh, n);
  n = 2 * NKV * HD * TC;k_f32_to_f16<<<(n + 255) / 256, 256, 0, stream>>>(wkv, wkvh, n);
  n = TC * TC;          k_f32_to_f16<<<(n + 255) / 256, 256, 0, stream>>>(wo, woh, n);

  k_gemm_awt<false><<<dim3(TC / 128, BT / 128), 256, 0, stream>>>(
      xh, wqh, qlin, nullptr, nullptr, TC, TC);
  k_gemm_awt<false><<<dim3((2 * NKV * HD) / 128, BT / 128), 256, 0, stream>>>(
      xh, wkvh, kvlin, nullptr, nullptr, 2 * NKV * HD, TC);

  n = TB * NH * TT * 64;
  k_rope_pack<<<(n + 255) / 256, 256, 0, stream>>>(qlin, fr, qh, NH, 0, TC);
  n = TB * NKV * TT * 64;
  k_rope_pack<<<(n + 255) / 256, 256, 0, stream>>>(kvlin, fr, kh, NKV, 0, 2 * NKV * HD);
  n = TB * NKV * TT * HD;
  k_pack_v<<<(n + 255) / 256, 256, 0, stream>>>(kvlin, vh);

  k_attn<<<dim3(TT / 64, TB * NH), 128, 0, stream>>>(qh, kh, vh, ao);

  k_gemm_awt<true><<<dim3(TC / 128, BT / 128), 256, 0, stream>>>(
      ao, woh, nullptr, out, bo, TC, TC);
}
